// FullQDisentangledVAE_76948634075858
// MI455X (gfx1250) — compile-verified, matrix operand order checked by reading, not run on hardware
//
#include <hip/hip_runtime.h>
#include <math.h>

// ---------------------------------------------------------------------------
// CDNA5 (gfx1250) wave32 WMMA types / helpers
// ---------------------------------------------------------------------------
typedef __attribute__((ext_vector_type(16))) __bf16 v16bf;
typedef __attribute__((ext_vector_type(8)))  float  v8f;

union BfVec { v16bf v; unsigned u[8]; unsigned short s[16]; };

__device__ __forceinline__ unsigned short f2bf(float f) {
    unsigned u = __builtin_bit_cast(unsigned, f);
    u += 0x7FFFu + ((u >> 16) & 1u);                 // round-to-nearest-even
    return (unsigned short)(u >> 16);
}
__device__ __forceinline__ float bf2f(unsigned short h) {
    unsigned u = ((unsigned)h) << 16;
    return __builtin_bit_cast(float, u);
}
__device__ __forceinline__ float sigf(float x) { return 1.f / (1.f + expf(-x)); }
__device__ __forceinline__ float softplusf(float x) {
    return (x > 20.f) ? x : log1pf(expf(x));
}

// ---------------------------------------------------------------------------
// Generic A-side loaders (dense + conv1)
// ---------------------------------------------------------------------------
struct ALDense {                 // bf16 row-major [M, ld], zero beyond kreal
    const unsigned short* p; int ld, kreal;
    __device__ void load8(int m, int k8, unsigned* d) const {
        if (k8 < kreal) {
            const uint4* q = (const uint4*)(p + (size_t)m * ld + k8);
            uint4 v = *q; d[0] = v.x; d[1] = v.y; d[2] = v.z; d[3] = v.w;
        } else d[0] = d[1] = d[2] = d[3] = 0u;
    }
};
struct ALConv {                  // implicit im2col, NHWC bf16, k = s*ICpad + ic
    const unsigned short* act; int icShift, IH, IW, owShift, ohowShift;
    __device__ void load8(int m, int k8, unsigned* d) const {
        d[0] = d[1] = d[2] = d[3] = 0u;
        int n  = m >> ohowShift, rem = m & ((1 << ohowShift) - 1);
        int oy = rem >> owShift, ox = rem & ((1 << owShift) - 1);
        int s  = k8 >> icShift,  icb = k8 & ((1 << icShift) - 1);
        int iy = 2 * oy - 1 + (s >> 2), ix = 2 * ox - 1 + (s & 3);
        if ((unsigned)iy >= (unsigned)IH || (unsigned)ix >= (unsigned)IW) return;
        const uint4* q = (const uint4*)(act + ((size_t)(((n * IH + iy) * IW) + ix) << icShift) + icb);
        uint4 v = *q; d[0] = v.x; d[1] = v.y; d[2] = v.z; d[3] = v.w;
    }
};

// ---------------------------------------------------------------------------
// Tap providers for the IC=256 conv / deconv implicit-GEMM kernel:
// decode + halo check ONCE per spatial tap (16 taps), 8 K-chunks per tap.
// ---------------------------------------------------------------------------
struct TapConv {                 // stride-2 pad-1 4x4 conv, NHWC bf16 (IC=256)
    const unsigned short* act; int IH, IW, owShift, ohowShift;
    __device__ const unsigned short* tap(int m, int s, bool& valid) const {
        int n  = m >> ohowShift, rem = m & ((1 << ohowShift) - 1);
        int oy = rem >> owShift, ox = rem & ((1 << owShift) - 1);
        int iy = 2 * oy - 1 + (s >> 2), ix = 2 * ox - 1 + (s & 3);
        valid = ((unsigned)iy < (unsigned)IH) && ((unsigned)ix < (unsigned)IW);
        return act + ((size_t)(((n * IH + iy) * IW) + ix) << 8);
    }
};
struct TapDeconv {               // lhs-dilation 2, pad 2, 4x4, NHWC bf16 (IC=256)
    const unsigned short* act; int IH, IW, owShift, ohowShift;
    __device__ const unsigned short* tap(int m, int s, bool& valid) const {
        int n  = m >> ohowShift, rem = m & ((1 << ohowShift) - 1);
        int oy = rem >> owShift, ox = rem & ((1 << owShift) - 1);
        int yd = oy - 2 + (s >> 2), xd = ox - 2 + (s & 3);
        int iy = yd >> 1, ix = xd >> 1;
        valid = !((yd | xd) < 0 || (yd & 1) || (xd & 1) || iy >= IH || ix >= IW);
        return act + ((size_t)(((n * IH + iy) * IW) + ix) << 8);
    }
};

// ---------------------------------------------------------------------------
// WMMA GEMM kernels: C[M,N] = act(A * W^T + bias), bf16 in / f32 acc,
// one 16x16 C tile per wave32, W pre-packed bf16 [tilesN*16, Kpad].
// ---------------------------------------------------------------------------
#define ACT_NONE  0
#define ACT_LRELU 1

__device__ __forceinline__ void store_tile(v8f acc, const float* bias, float* Cf,
                                           unsigned short* Cb, int M, int N,
                                           int m0, int n0, int row, int half, int act)
{
#pragma unroll
    for (int r = 0; r < 8; ++r) {                    // D: n = lane%16, m = r + 8*half
        int mm = m0 + r + 8 * half;
        int nn = n0 + row;
        if (mm < M && nn < N) {
            float v = acc[r];
            if (bias) v += bias[nn];
            if (act == ACT_LRELU) v = v > 0.f ? v : 0.1f * v;
            if (Cf) Cf[(size_t)mm * N + nn] = v;
            else    Cb[(size_t)mm * N + nn] = f2bf(v);
        }
    }
}

template <typename AL>
__global__ __launch_bounds__(128)
void gemm_wmma(AL A, const unsigned short* Bw, int ldb, const float* bias,
               float* Cf, unsigned short* Cb, int M, int N, int Kpad,
               int act, int tilesN, int tiles)
{
    int wave = threadIdx.x >> 5;
    int tile = blockIdx.x * 4 + wave;
    if (tile >= tiles) return;
    int tm = tile / tilesN, tn = tile - tm * tilesN;
    int m0 = tm * 16, n0 = tn * 16;
    int lane = threadIdx.x & 31;
    int row  = lane & 15;
    int half = lane >> 4;
    int m = m0 + row;
    const unsigned short* brow = Bw + (size_t)(n0 + row) * ldb;

    v8f acc = {0.f, 0.f, 0.f, 0.f, 0.f, 0.f, 0.f, 0.f};
    for (int kb = 0; kb < Kpad; kb += 32) {
        int ka = kb + half * 8;                      // ISA 16-bit A/B K pattern
        BfVec a, b;
        A.load8(m, ka,      &a.u[0]);
        A.load8(m, ka + 16, &a.u[4]);
        uint4 b0 = *(const uint4*)(brow + ka);
        uint4 b1 = *(const uint4*)(brow + ka + 16);
        b.u[0] = b0.x; b.u[1] = b0.y; b.u[2] = b0.z; b.u[3] = b0.w;
        b.u[4] = b1.x; b.u[5] = b1.y; b.u[6] = b1.z; b.u[7] = b1.w;
        if (kb + 128 < Kpad) __builtin_prefetch(brow + kb + 128, 0, 1);
        acc = __builtin_amdgcn_wmma_f32_16x16x32_bf16(
            false, a.v, false, b.v, (short)0, acc, false, false);
    }
    store_tile(acc, bias, Cf, Cb, M, N, m0, n0, row, half, act);
}

// Tap-outer / channel-inner implicit GEMM for IC=256 conv/deconv (Kpad = 4096)
template <typename AL>
__global__ __launch_bounds__(128)
void gemm_wmma_tap(AL A, const unsigned short* Bw, const float* bias,
                   float* Cf, unsigned short* Cb, int M, int N,
                   int act, int tilesN, int tiles)
{
    int wave = threadIdx.x >> 5;
    int tile = blockIdx.x * 4 + wave;
    if (tile >= tiles) return;
    int tm = tile / tilesN, tn = tile - tm * tilesN;
    int m0 = tm * 16, n0 = tn * 16;
    int lane = threadIdx.x & 31;
    int row  = lane & 15;
    int half = lane >> 4;
    int m = m0 + row;
    const unsigned short* brow = Bw + ((size_t)(n0 + row) << 12);

    v8f acc = {0.f, 0.f, 0.f, 0.f, 0.f, 0.f, 0.f, 0.f};
    for (int s = 0; s < 16; ++s) {                   // 16 spatial taps
        bool valid;
        const unsigned short* ab = A.tap(m, s, valid);
        const unsigned short* bb = brow + (s << 8);
        if (s < 15) __builtin_prefetch(bb + 256, 0, 1);
#pragma unroll
        for (int icb = 0; icb < 256; icb += 32) {    // 8 K-chunks per tap
            int ka = icb + half * 8;
            BfVec a, b;
            if (valid) {
                uint4 a0 = *(const uint4*)(ab + ka);
                uint4 a1 = *(const uint4*)(ab + ka + 16);
                a.u[0] = a0.x; a.u[1] = a0.y; a.u[2] = a0.z; a.u[3] = a0.w;
                a.u[4] = a1.x; a.u[5] = a1.y; a.u[6] = a1.z; a.u[7] = a1.w;
            } else {
                a.u[0] = a.u[1] = a.u[2] = a.u[3] = 0u;
                a.u[4] = a.u[5] = a.u[6] = a.u[7] = 0u;
            }
            uint4 b0 = *(const uint4*)(bb + ka);
            uint4 b1 = *(const uint4*)(bb + ka + 16);
            b.u[0] = b0.x; b.u[1] = b0.y; b.u[2] = b0.z; b.u[3] = b0.w;
            b.u[4] = b1.x; b.u[5] = b1.y; b.u[6] = b1.z; b.u[7] = b1.w;
            acc = __builtin_amdgcn_wmma_f32_16x16x32_bf16(
                false, a.v, false, b.v, (short)0, acc, false, false);
        }
    }
    store_tile(acc, bias, Cf, Cb, M, N, m0, n0, row, half, act);
}

template <typename AL>
static void gemmB(AL a, const unsigned short* Bw, int ldb, const float* bias,
                  float* Cf, unsigned short* Cb, int M, int N, int Kpad, int act,
                  hipStream_t s)
{
    int tilesN = (N + 15) / 16;
    int tiles  = ((M + 15) / 16) * tilesN;
    gemm_wmma<AL><<<(tiles + 3) / 4, 128, 0, s>>>(a, Bw, ldb, bias, Cf, Cb,
                                                  M, N, Kpad, act, tilesN, tiles);
}
template <typename AL>
static void gemmTap(AL a, const unsigned short* Bw, const float* bias,
                    float* Cf, unsigned short* Cb, int M, int N, int act,
                    hipStream_t s)
{
    int tilesN = (N + 15) / 16;
    int tiles  = ((M + 15) / 16) * tilesN;
    gemm_wmma_tap<AL><<<(tiles + 3) / 4, 128, 0, s>>>(a, Bw, bias, Cf, Cb,
                                                      M, N, act, tilesN, tiles);
}

// ---------------------------------------------------------------------------
// Weight / input packing (f32 -> bf16, K reordered to match A-side traversal)
// ---------------------------------------------------------------------------
__global__ void pack_dense_kernel(const float* w, unsigned short* o,
                                  int N, int K, int Kpad, int total) {
    int i = blockIdx.x * blockDim.x + threadIdx.x;
    if (i >= total) return;
    int n = i / Kpad, k = i - n * Kpad;
    float v = (n < N && k < K) ? w[(size_t)n * K + k] : 0.f;
    o[i] = f2bf(v);
}
__global__ void pack_convw_kernel(const float* w, unsigned short* o,
                                  int IC, int icpShift, int Kpad, int total) {
    // k-order: s*ICpad + ic   (w layout: [OC][IC][4][4]); ic >= IC zero-padded
    int i = blockIdx.x * blockDim.x + threadIdx.x;
    if (i >= total) return;
    int oc = i / Kpad, k = i - oc * Kpad;
    float v = 0.f;
    if (k < (16 << icpShift)) {
        int s = k >> icpShift, ic = k & ((1 << icpShift) - 1);
        if (ic < IC) v = w[(size_t)(oc * IC + ic) * 16 + s];
    }
    o[i] = f2bf(v);
}
__global__ void pack_deconvw_kernel(const float* w, unsigned short* o,
                                    int IC, int OC, int Kpad, int total) {
    // transpose + spatial flip folded in (w layout: [IC][OC][4][4]); rows padded
    int i = blockIdx.x * blockDim.x + threadIdx.x;
    if (i >= total) return;
    int oc = i / Kpad, k = i - oc * Kpad;
    float v = 0.f;
    if (oc < OC && k < IC * 16) {
        int s = k / IC, ic = k - s * IC, ky = s >> 2, kx = s & 3;
        v = w[(size_t)((ic * OC + oc) * 4 + (3 - ky)) * 4 + (3 - kx)];
    }
    o[i] = f2bf(v);
}
__global__ void pack_fcw_kernel(const float* w, unsigned short* o) {
    // conv_fc_w (512,4096): reorder K from NCHW-flat (c*16+pos) to NHWC (pos*256+c)
    int i = blockIdx.x * blockDim.x + threadIdx.x;
    if (i >= 512 * 4096) return;
    int n = i >> 12, k = i & 4095;
    int pos = k >> 8, c = k & 255;
    o[i] = f2bf(w[(size_t)n * 4096 + c * 16 + pos]);
}
// x (128,3,64,64) f32 NCHW -> NHWC bf16 with IC padded 3->8
__global__ void cvt_x_kernel(const float* x, unsigned short* o) {
    int i = blockIdx.x * blockDim.x + threadIdx.x;
    if (i >= 128 * 4096 * 8) return;
    int pix = i >> 3, ic = i & 7;
    int n = pix >> 12, pos = pix & 4095;
    float v = 0.f;
    if (ic < 3) v = x[((size_t)(n * 3 + ic) << 12) + pos];
    o[i] = f2bf(v);
}
__global__ void cvt_bf16_kernel(const float* x, unsigned short* o, int n) {
    int i = blockIdx.x * blockDim.x + threadIdx.x;
    if (i < n) o[i] = f2bf(x[i]);
}

// ---------------------------------------------------------------------------
// Elementwise / reduction kernels
// ---------------------------------------------------------------------------
__global__ void fill_kernel(float* p, float v, int n) {
    int i = blockIdx.x * blockDim.x + threadIdx.x;
    if (i < n) p[i] = v;
}
__global__ void fillh_kernel(unsigned short* p, unsigned short v, int n) {
    int i = blockIdx.x * blockDim.x + threadIdx.x;
    if (i < n) p[i] = v;
}
// coalesced column sums: threads <-> columns, row-slab per block, atomic partials
__global__ void colsum_kernel(const float* x, int M, int N, int rowsPerBlock,
                              float* sum, float* sq) {
    int r0 = blockIdx.x * rowsPerBlock;
    int r1 = r0 + rowsPerBlock; if (r1 > M) r1 = M;
    for (int n = threadIdx.x; n < N; n += blockDim.x) {
        float s = 0.f, q = 0.f;
        for (int m = r0; m < r1; ++m) {
            float v = x[(size_t)m * N + n];
            s += v; q += v * v;
        }
        atomicAdd(&sum[n], s);
        atomicAdd(&sq[n], q);
    }
}
__global__ void colfin_kernel(const float* sum, const float* sq, int N, float invM,
                              float* mean, float* var) {
    int i = blockIdx.x * blockDim.x + threadIdx.x;
    if (i >= N) return;
    float mu = sum[i] * invM;
    mean[i] = mu;
    var[i]  = sq[i] * invM - mu * mu;
}
// BN + lrelu, writes bf16 activation for next WMMA stage (N = power of two)
__global__ void bn_apply_bf16_kernel(const float* x, unsigned short* y, int total, int nMask,
                                     const float* mean, const float* var,
                                     const float* g, const float* b) {
    int i = blockIdx.x * blockDim.x + threadIdx.x;
    if (i >= total) return;
    int n = i & nMask;
    float v = g[n] * (x[i] - mean[n]) * rsqrtf(var[n] + 1e-5f) + b[n];
    v = v > 0.f ? v : 0.1f * v;
    y[i] = f2bf(v);
}
// dBN-fc variant: input [128, 4096] with k = c*16+pos; output NHWC bf16 (n*16+pos, c)
__global__ void bn_apply_perm_kernel(const float* x, unsigned short* y,
                                     const float* mean, const float* var,
                                     const float* g, const float* b) {
    int i = blockIdx.x * blockDim.x + threadIdx.x;
    if (i >= 128 * 4096) return;
    int n = i >> 12, k = i & 4095;
    int c = k >> 4, pos = k & 15;
    float v = g[k] * (x[i] - mean[k]) * rsqrtf(var[k] + 1e-5f) + b[k];
    v = v > 0.f ? v : 0.1f * v;
    y[(size_t)(n * 16 + pos) * 256 + c] = f2bf(v);
}
// gates order: i, f, g, o (chunks of 512); h/cat in bf16, c in f32
__global__ void lstm_cell_kernel(const float* xw, const float* gh,
                                 const float* bih, const float* bhh,
                                 unsigned short* h, float* c, unsigned short* cat,
                                 int t, int coloff) {
    int i = blockIdx.x * blockDim.x + threadIdx.x;
    if (i >= 16 * 512) return;
    int b = i >> 9, j = i & 511;
    const float* xr = xw + (size_t)(b * 8 + t) * 2048;   // rows are n = b*T + t
    const float* gr = gh + (size_t)b * 2048;
    float gi = xr[j]        + gr[j]        + bih[j]        + bhh[j];
    float gf = xr[512 + j]  + gr[512 + j]  + bih[512 + j]  + bhh[512 + j];
    float gg = xr[1024 + j] + gr[1024 + j] + bih[1024 + j] + bhh[1024 + j];
    float go = xr[1536 + j] + gr[1536 + j] + bih[1536 + j] + bhh[1536 + j];
    float cn = sigf(gf) * c[i] + sigf(gi) * tanhf(gg);
    float hn = sigf(go) * tanhf(cn);
    c[i] = cn;
    unsigned short hb = f2bf(hn);
    h[i] = hb;
    cat[(size_t)(t * 16 + b) * 1024 + coloff + j] = hb;
}
__global__ void rnn_cell_kernel(const float* xw, const float* gh,
                                const float* bih, const float* bhh,
                                unsigned short* h, unsigned short* hout) {
    int i = blockIdx.x * blockDim.x + threadIdx.x;
    if (i >= 16 * 512) return;
    int j = i & 511;
    unsigned short v = f2bf(tanhf(xw[i] + gh[i] + bih[j] + bhh[j]));
    h[i] = v; hout[i] = v;
}
__global__ void post_kernel(const float* mlin, const float* slin, const float* eps,
                            float* pm, float* ps, float* zbase, int zrowstride,
                            unsigned short* zdec, int zdrowstride) {
    int i = blockIdx.x * blockDim.x + threadIdx.x;
    if (i >= 16 * 144) return;
    int b = i / 144, d = i - b * 144;
    float m = mlin[i];
    float s = softplusf(slin[i]) + 1e-5f;
    float z = m + s * eps[i];
    pm[i] = m; ps[i] = s;
    zbase[(size_t)b * zrowstride + d] = z;
    zdec[(size_t)b * zdrowstride + d] = f2bf(z);
}
__global__ void prior_kernel(const float* mlin, const float* slin, const float* eps,
                             float* qm, float* qs, unsigned short* zt) {
    int i = blockIdx.x * blockDim.x + threadIdx.x;
    if (i >= 16 * 144) return;
    float m = mlin[i];
    float s = softplusf(slin[i]) + 1e-5f;
    qm[i] = m; qs[i] = s;
    zt[i] = f2bf(m + s * eps[i]);
}
__global__ void gru_pre_kernel(const unsigned short* zt, unsigned short* tmp0,
                               unsigned short* tmp2) {
    int i = blockIdx.x * blockDim.x + threadIdx.x;
    if (i >= 16 * 144) return;
    int d = i % 144;
    tmp0[i] = (d < 96) ? zt[i] : (unsigned short)0;
    tmp2[i] = (d >= 48) ? zt[i] : (unsigned short)0;
}
// gi includes bih, gh includes bhh (GEMM epilogues); h state in bf16
__global__ void gru_cell_kernel(const float* gi, const float* gh,
                                unsigned short* h, const float* wt, int wcol) {
    int i = blockIdx.x * blockDim.x + threadIdx.x;
    if (i >= 16 * 48) return;
    int b = i / 48, e = i - b * 48;
    float r  = sigf(gi[b * 144 + e]      + gh[b * 144 + e]);
    float zg = sigf(gi[b * 144 + 48 + e] + gh[b * 144 + 48 + e]);
    float nn = tanhf(gi[b * 144 + 96 + e] + r * gh[b * 144 + 96 + e]);
    float ho = bf2f(h[i]);
    float hnew = (1.f - zg) * nn + zg * ho;
    float w = wt[b * 3 + wcol];
    h[i] = f2bf(w * hnew + (1.f - w) * ho);
}
__global__ void interleave_kernel(const unsigned short* z0, const unsigned short* z1,
                                  const unsigned short* z2, unsigned short* zfa) {
    int i = blockIdx.x * blockDim.x + threadIdx.x;
    if (i >= 16 * 144) return;
    int b = i / 144, q = i - b * 144;
    int e = q / 3, j = q - e * 3;
    const unsigned short* src = (j == 0) ? z0 : (j == 1) ? z1 : z2;
    zfa[i] = src[b * 48 + e];
}
__global__ void wt_kernel(const unsigned short* zfa, const float* zww,
                          const float* zwb, float* wt) {
    int b = threadIdx.x;
    if (b >= 16) return;
    float l[3];
    for (int j = 0; j < 3; ++j) {
        float s = zwb[j];
        for (int k = 0; k < 144; ++k) s += bf2f(zfa[b * 144 + k]) * zww[j * 144 + k];
        l[j] = s;
    }
    float m  = fmaxf(l[0], fmaxf(l[1], l[2]));
    float e0 = expf(l[0] - m), e1 = expf(l[1] - m), e2 = expf(l[2] - m);
    float inv = 1.f / (e0 + e1 + e2);
    float p0 = e0 * inv, p1 = e1 * inv;
    wt[b * 3 + 0] = p0;
    wt[b * 3 + 1] = p0 + p1;
    wt[b * 3 + 2] = p0 + p1 + e2 * inv;
}
// y: [128*4096 rows (n,oy,ox), 3 cols] f32 -> recon (B,T,3,64,64) NCHW with tanh
__global__ void recon_kernel(const float* y, float* out) {
    int i = blockIdx.x * blockDim.x + threadIdx.x;
    if (i >= 128 * 4096 * 3) return;
    int m = i / 3, cc = i - m * 3;
    int n = m >> 12, pos = m & 4095;
    out[(size_t)n * 12288 + (size_t)cc * 4096 + pos] = tanhf(y[i]);
}

// ---------------------------------------------------------------------------
// Input index map (setup_inputs dict order; params dict in insertion order)
// ---------------------------------------------------------------------------
enum {
    IN_X = 0,
    IN_CONV1_W, IN_CONV2_W, IN_CONV3_W, IN_CONV4_W, IN_CONV_FC_W,
    IN_LSTM_WIH_F, IN_LSTM_WHH_F, IN_LSTM_WIH_B, IN_LSTM_WHH_B,
    IN_RNN_WIH, IN_RNN_WHH,
    IN_Z_MEAN_W, IN_Z_LOGVAR_W, IN_Z_MEAN_PRIOR_W, IN_Z_LOGVAR_PRIOR_W,
    IN_GRU0_WIH, IN_GRU0_WHH, IN_GRU1_WIH, IN_GRU1_WHH, IN_GRU2_WIH, IN_GRU2_WHH,
    IN_Z_W_W, IN_DECONV_FC_W, IN_DECONV4_W, IN_DECONV3_W, IN_DECONV2_W, IN_DECONV1_W,
    IN_CONV1_B, IN_CONV2_B, IN_CONV3_B, IN_CONV4_B, IN_CONV_FC_B,
    IN_LSTM_BIH_F, IN_LSTM_BHH_F, IN_LSTM_BIH_B, IN_LSTM_BHH_B,
    IN_RNN_BIH, IN_RNN_BHH,
    IN_Z_MEAN_B, IN_Z_LOGVAR_B, IN_Z_MEAN_PRIOR_B, IN_Z_LOGVAR_PRIOR_B,
    IN_GRU0_BIH, IN_GRU0_BHH, IN_GRU1_BIH, IN_GRU1_BHH, IN_GRU2_BIH, IN_GRU2_BHH,
    IN_Z_W_B, IN_DECONV_FC_B, IN_DECONV4_B, IN_DECONV3_B, IN_DECONV2_B, IN_DECONV1_B,
    IN_BN2_G, IN_BN2_B, IN_BN3_G, IN_BN3_B, IN_BN4_G, IN_BN4_B,
    IN_DBN4_G, IN_DBN4_B, IN_DBN3_G, IN_DBN3_B, IN_DBN2_G, IN_DBN2_B,
    IN_BNF_G, IN_BNF_B, IN_DBNF_G, IN_DBNF_B,
    IN_NOISE_POST, IN_NOISE_PRIOR, IN_EPS0
};

static inline void fill(float* p, float v, int n, hipStream_t s) {
    fill_kernel<<<(n + 255) / 256, 256, 0, s>>>(p, v, n);
}
static inline void fillh(unsigned short* p, unsigned short v, int n, hipStream_t s) {
    fillh_kernel<<<(n + 255) / 256, 256, 0, s>>>(p, v, n);
}

extern "C" void kernel_launch(void* const* d_in, const int* in_sizes, int n_in,
                              void* d_out, int out_size, void* d_ws, size_t ws_size,
                              hipStream_t stream)
{
    (void)in_sizes; (void)n_in; (void)out_size; (void)ws_size;
    auto F = [&](int i) { return (const float*)d_in[i]; };
    typedef unsigned short u16;

    // ---- outputs -----------------------------------------------------------
    float* out  = (float*)d_out;
    const int TBZ = 8 * 16 * 144;           // 18432
    float* pm_o = out;
    float* ps_o = out + TBZ;
    float* qm_o = out + 2 * TBZ;
    float* qs_o = out + 3 * TBZ;
    float* z_o  = out + 4 * TBZ;            // (B,T,ZD)
    float* rc_o = out + 5 * TBZ;            // recon (B,T,3,64,64)

    // ---- workspace bump allocator (64B aligned) ---------------------------
    char* base = (char*)d_ws;
    size_t off = 0;
    auto allocRaw = [&](size_t bytes) {
        off = (off + 63) & ~(size_t)63;
        void* p = base + off; off += bytes; return p;
    };
    auto allocF = [&](size_t n) { return (float*)allocRaw(n * 4); };
    auto allocH = [&](size_t n) { return (u16*)allocRaw(n * 2); };

    // packed bf16 weights
    u16* wc1  = allocH(256 * 128);
    u16* wc2  = allocH((size_t)256 * 4096);
    u16* wc3  = allocH((size_t)256 * 4096);
    u16* wc4  = allocH((size_t)256 * 4096);
    u16* wfc  = allocH((size_t)512 * 4096);
    u16* wlihF = allocH((size_t)2048 * 512);
    u16* wlhhF = allocH((size_t)2048 * 512);
    u16* wlihB = allocH((size_t)2048 * 512);
    u16* wlhhB = allocH((size_t)2048 * 512);
    u16* wrih = allocH((size_t)512 * 1024);
    u16* wrhh = allocH((size_t)512 * 512);
    u16* wzm  = allocH(144 * 512);
    u16* wzlv = allocH(144 * 512);
    u16* wzmp = allocH(144 * 160);
    u16* wzlp = allocH(144 * 160);
    u16* wg0i = allocH(144 * 160); u16* wg0h = allocH(144 * 64);
    u16* wg1i = allocH(144 * 160); u16* wg1h = allocH(144 * 64);
    u16* wg2i = allocH(144 * 160); u16* wg2h = allocH(144 * 64);
    u16* wdfc = allocH((size_t)4096 * 160);
    u16* wd4  = allocH((size_t)256 * 4096);
    u16* wd3  = allocH((size_t)256 * 4096);
    u16* wd2  = allocH((size_t)256 * 4096);
    u16* wd1  = allocH(16 * 4096);

    // f32 scratch
    float* mean  = allocF(4096);
    float* var   = allocF(4096);
    float* ssum  = allocF(4096);
    float* ssq   = allocF(4096);
    float* hfcF  = allocF(128 * 512);
    float* xwF   = allocF((size_t)128 * 2048);
    float* xwB   = allocF((size_t)128 * 2048);
    float* xwR   = allocF(128 * 512);
    float* ghbuf = allocF(16 * 2048);
    float* cF = allocF(16 * 512); float* cB = allocF(16 * 512);
    float* mlin = allocF(16 * 144); float* slin = allocF(16 * 144);
    float* gi   = allocF(16 * 144); float* ghg  = allocF(16 * 144);
    float* wtb  = allocF(64);

    // bf16 activations / states
    u16* xbf   = allocH((size_t)128 * 4096 * 8);    // input NHWC, IC padded to 8
    u16* hfc   = allocH(128 * 512);
    u16* cat   = allocH((size_t)128 * 1024);
    u16* hF = allocH(16 * 512); u16* hB = allocH(16 * 512); u16* hR = allocH(16 * 512);
    u16* rnnOut = allocH(128 * 512);
    u16* zt   = allocH(16 * 144);
    u16* tmp0 = allocH(16 * 144); u16* tmp2 = allocH(16 * 144);
    u16* zfa  = allocH(16 * 144);
    u16* zf0 = allocH(16 * 48); u16* zf1 = allocH(16 * 48); u16* zf2 = allocH(16 * 48);
    u16* zdec = allocH(128 * 144);

    // big buffers
    float* bufF = allocF((size_t)131072 * 256);     // f32 GEMM out for BN stats
    u16* bfP = allocH((size_t)131072 * 256);        // bf16 activation ping
    u16* bfQ = allocH((size_t)131072 * 256);        // bf16 activation pong

    // BN stats helper: zero partials, coalesced slab sums, finalize
    auto bnstats = [&](const float* x, int M, int N) {
        fill(ssum, 0.f, N, stream);
        fill(ssq,  0.f, N, stream);
        int rows = 128, blocks = (M + rows - 1) / rows;
        colsum_kernel<<<blocks, 256, 0, stream>>>(x, M, N, rows, ssum, ssq);
        colfin_kernel<<<(N + 255) / 256, 256, 0, stream>>>(ssum, ssq, N, 1.f / (float)M, mean, var);
    };

    // ======================= Weight / input packing ========================
    auto packD = [&](const float* w, u16* o, int N, int K, int Kpad) {
        int total = N * Kpad;
        pack_dense_kernel<<<(total + 255) / 256, 256, 0, stream>>>(w, o, N, K, Kpad, total);
    };
    cvt_x_kernel<<<(128 * 4096 * 8 + 255) / 256, 256, 0, stream>>>(F(IN_X), xbf);
    pack_convw_kernel<<<(256 * 128 + 255) / 256, 256, 0, stream>>>(F(IN_CONV1_W), wc1, 3, 3, 128, 256 * 128);
    pack_convw_kernel<<<(256 * 4096 + 255) / 256, 256, 0, stream>>>(F(IN_CONV2_W), wc2, 256, 8, 4096, 256 * 4096);
    pack_convw_kernel<<<(256 * 4096 + 255) / 256, 256, 0, stream>>>(F(IN_CONV3_W), wc3, 256, 8, 4096, 256 * 4096);
    pack_convw_kernel<<<(256 * 4096 + 255) / 256, 256, 0, stream>>>(F(IN_CONV4_W), wc4, 256, 8, 4096, 256 * 4096);
    pack_fcw_kernel<<<(512 * 4096 + 255) / 256, 256, 0, stream>>>(F(IN_CONV_FC_W), wfc);
    packD(F(IN_LSTM_WIH_F), wlihF, 2048, 512, 512);
    packD(F(IN_LSTM_WHH_F), wlhhF, 2048, 512, 512);
    packD(F(IN_LSTM_WIH_B), wlihB, 2048, 512, 512);
    packD(F(IN_LSTM_WHH_B), wlhhB, 2048, 512, 512);
    packD(F(IN_RNN_WIH), wrih, 512, 1024, 1024);
    packD(F(IN_RNN_WHH), wrhh, 512, 512, 512);
    packD(F(IN_Z_MEAN_W), wzm, 144, 512, 512);
    packD(F(IN_Z_LOGVAR_W), wzlv, 144, 512, 512);
    packD(F(IN_Z_MEAN_PRIOR_W), wzmp, 144, 144, 160);
    packD(F(IN_Z_LOGVAR_PRIOR_W), wzlp, 144, 144, 160);
    packD(F(IN_GRU0_WIH), wg0i, 144, 144, 160); packD(F(IN_GRU0_WHH), wg0h, 144, 48, 64);
    packD(F(IN_GRU1_WIH), wg1i, 144, 144, 160); packD(F(IN_GRU1_WHH), wg1h, 144, 48, 64);
    packD(F(IN_GRU2_WIH), wg2i, 144, 144, 160); packD(F(IN_GRU2_WHH), wg2h, 144, 48, 64);
    packD(F(IN_DECONV_FC_W), wdfc, 4096, 144, 160);
    pack_deconvw_kernel<<<(256 * 4096 + 255) / 256, 256, 0, stream>>>(F(IN_DECONV4_W), wd4, 256, 256, 4096, 256 * 4096);
    pack_deconvw_kernel<<<(256 * 4096 + 255) / 256, 256, 0, stream>>>(F(IN_DECONV3_W), wd3, 256, 256, 4096, 256 * 4096);
    pack_deconvw_kernel<<<(256 * 4096 + 255) / 256, 256, 0, stream>>>(F(IN_DECONV2_W), wd2, 256, 256, 4096, 256 * 4096);
    pack_deconvw_kernel<<<(16 * 4096 + 255) / 256, 256, 0, stream>>>(F(IN_DECONV1_W), wd1, 256, 3, 4096, 16 * 4096);

    // ======================= Encoder =======================================
    // conv1 (ICpad=8, K=128): generic path -> bf16 NHWC [131072,256], bias+lrelu
    gemmB(ALConv{xbf, 3, 64, 64, 5, 10}, wc1, 128, F(IN_CONV1_B),
          nullptr, bfP, 131072, 256, 128, ACT_LRELU, stream);
    // conv2..4: tap-structured implicit GEMM (IC=256)
    gemmTap(TapConv{bfP, 32, 32, 4, 8}, wc2, F(IN_CONV2_B),
            bufF, nullptr, 32768, 256, ACT_NONE, stream);
    bnstats(bufF, 32768, 256);
    bn_apply_bf16_kernel<<<(32768 * 256 + 255) / 256, 256, 0, stream>>>(
        bufF, bfQ, 32768 * 256, 255, mean, var, F(IN_BN2_G), F(IN_BN2_B));
    gemmTap(TapConv{bfQ, 16, 16, 3, 6}, wc3, F(IN_CONV3_B),
            bufF, nullptr, 8192, 256, ACT_NONE, stream);
    bnstats(bufF, 8192, 256);
    bn_apply_bf16_kernel<<<(8192 * 256 + 255) / 256, 256, 0, stream>>>(
        bufF, bfP, 8192 * 256, 255, mean, var, F(IN_BN3_G), F(IN_BN3_B));
    gemmTap(TapConv{bfP, 8, 8, 2, 4}, wc4, F(IN_CONV4_B),
            bufF, nullptr, 2048, 256, ACT_NONE, stream);
    bnstats(bufF, 2048, 256);
    bn_apply_bf16_kernel<<<(2048 * 256 + 255) / 256, 256, 0, stream>>>(
        bufF, bfQ, 2048 * 256, 255, mean, var, F(IN_BN4_G), F(IN_BN4_B));
    // conv_fc (K reordered to NHWC in wfc) -> f32; BN1d -> bf16 hfc
    gemmB(ALDense{bfQ, 4096, 4096}, wfc, 4096, F(IN_CONV_FC_B),
          hfcF, nullptr, 128, 512, 4096, ACT_NONE, stream);
    bnstats(hfcF, 128, 512);
    bn_apply_bf16_kernel<<<(128 * 512 + 255) / 256, 256, 0, stream>>>(
        hfcF, hfc, 128 * 512, 511, mean, var, F(IN_BNF_G), F(IN_BNF_B));

    // ======================= Bi-LSTM + RNN =================================
    gemmB(ALDense{hfc, 512, 512}, wlihF, 512, nullptr, xwF, nullptr, 128, 2048, 512, ACT_NONE, stream);
    gemmB(ALDense{hfc, 512, 512}, wlihB, 512, nullptr, xwB, nullptr, 128, 2048, 512, ACT_NONE, stream);
    fillh(hF, 0, 16 * 512, stream); fill(cF, 0.f, 16 * 512, stream);
    fillh(hB, 0, 16 * 512, stream); fill(cB, 0.f, 16 * 512, stream);
    for (int s = 0; s < 8; ++s) {
        gemmB(ALDense{hF, 512, 512}, wlhhF, 512, nullptr, ghbuf, nullptr, 16, 2048, 512, ACT_NONE, stream);
        lstm_cell_kernel<<<32, 256, 0, stream>>>(xwF, ghbuf, F(IN_LSTM_BIH_F), F(IN_LSTM_BHH_F),
                                                 hF, cF, cat, s, 0);
        gemmB(ALDense{hB, 512, 512}, wlhhB, 512, nullptr, ghbuf, nullptr, 16, 2048, 512, ACT_NONE, stream);
        lstm_cell_kernel<<<32, 256, 0, stream>>>(xwB, ghbuf, F(IN_LSTM_BIH_B), F(IN_LSTM_BHH_B),
                                                 hB, cB, cat, 7 - s, 512);
    }
    gemmB(ALDense{cat, 1024, 1024}, wrih, 1024, nullptr, xwR, nullptr, 128, 512, 1024, ACT_NONE, stream);
    fillh(hR, 0, 16 * 512, stream);
    for (int t = 0; t < 8; ++t) {
        gemmB(ALDense{hR, 512, 512}, wrhh, 512, nullptr, ghbuf, nullptr, 16, 512, 512, ACT_NONE, stream);
        rnn_cell_kernel<<<32, 256, 0, stream>>>(xwR + (size_t)t * 16 * 512, ghbuf,
                                                F(IN_RNN_BIH), F(IN_RNN_BHH),
                                                hR, rnnOut + (size_t)t * 16 * 512);
    }

    // ======================= Latent scan ===================================
    const float* npost  = F(IN_NOISE_POST);
    const float* nprior = F(IN_NOISE_PRIOR);
    gemmB(ALDense{rnnOut, 512, 512}, wzm, 512, F(IN_Z_MEAN_B), mlin, nullptr, 16, 144, 512, ACT_NONE, stream);
    gemmB(ALDense{rnnOut, 512, 512}, wzlv, 512, F(IN_Z_LOGVAR_B), slin, nullptr, 16, 144, 512, ACT_NONE, stream);
    post_kernel<<<9, 256, 0, stream>>>(mlin, slin, npost, pm_o, ps_o, z_o, 8 * 144, zdec, 8 * 144);
    fill(qm_o, 0.f, 16 * 144, stream);
    fill(qs_o, 1.f, 16 * 144, stream);
    cvt_bf16_kernel<<<9, 256, 0, stream>>>(F(IN_EPS0), zt, 16 * 144);
    fill(wtb, 1.f, 48, stream);
    fillh(zf0, 0, 16 * 48, stream); fillh(zf1, 0, 16 * 48, stream); fillh(zf2, 0, 16 * 48, stream);

    for (int t = 1; t < 8; ++t) {
        const u16* lt = rnnOut + (size_t)t * 16 * 512;
        gemmB(ALDense{lt, 512, 512}, wzm, 512, F(IN_Z_MEAN_B), mlin, nullptr, 16, 144, 512, ACT_NONE, stream);
        gemmB(ALDense{lt, 512, 512}, wzlv, 512, F(IN_Z_LOGVAR_B), slin, nullptr, 16, 144, 512, ACT_NONE, stream);
        post_kernel<<<9, 256, 0, stream>>>(mlin, slin, npost + (size_t)t * 16 * 144,
                                           pm_o + (size_t)t * 16 * 144,
                                           ps_o + (size_t)t * 16 * 144,
                                           z_o + (size_t)t * 144, 8 * 144,
                                           zdec + (size_t)t * 144, 8 * 144);
        gru_pre_kernel<<<9, 256, 0, stream>>>(zt, tmp0, tmp2);
        gemmB(ALDense{tmp0, 144, 144}, wg0i, 160, F(IN_GRU0_BIH), gi, nullptr, 16, 144, 160, ACT_NONE, stream);
        gemmB(ALDense{zf0, 48, 48},    wg0h, 64,  F(IN_GRU0_BHH), ghg, nullptr, 16, 144, 64, ACT_NONE, stream);
        gru_cell_kernel<<<3, 256, 0, stream>>>(gi, ghg, zf0, wtb, 0);
        gemmB(ALDense{zt, 144, 144},   wg1i, 160, F(IN_GRU1_BIH), gi, nullptr, 16, 144, 160, ACT_NONE, stream);
        gemmB(ALDense{zf1, 48, 48},    wg1h, 64,  F(IN_GRU1_BHH), ghg, nullptr, 16, 144, 64, ACT_NONE, stream);
        gru_cell_kernel<<<3, 256, 0, stream>>>(gi, ghg, zf1, wtb, 1);
        gemmB(ALDense{tmp2, 144, 144}, wg2i, 160, F(IN_GRU2_BIH), gi, nullptr, 16, 144, 160, ACT_NONE, stream);
        gemmB(ALDense{zf2, 48, 48},    wg2h, 64,  F(IN_GRU2_BHH), ghg, nullptr, 16, 144, 64, ACT_NONE, stream);
        gru_cell_kernel<<<3, 256, 0, stream>>>(gi, ghg, zf2, wtb, 2);
        interleave_kernel<<<9, 256, 0, stream>>>(zf0, zf1, zf2, zfa);
        wt_kernel<<<1, 32, 0, stream>>>(zfa, F(IN_Z_W_W), F(IN_Z_W_B), wtb);
        gemmB(ALDense{zfa, 144, 144}, wzmp, 160, F(IN_Z_MEAN_PRIOR_B), mlin, nullptr, 16, 144, 160, ACT_NONE, stream);
        gemmB(ALDense{zfa, 144, 144}, wzlp, 160, F(IN_Z_LOGVAR_PRIOR_B), slin, nullptr, 16, 144, 160, ACT_NONE, stream);
        prior_kernel<<<9, 256, 0, stream>>>(mlin, slin, nprior + (size_t)(t - 1) * 16 * 144,
                                            qm_o + (size_t)t * 16 * 144,
                                            qs_o + (size_t)t * 16 * 144, zt);
    }

    // ======================= Decoder =======================================
    gemmB(ALDense{zdec, 144, 144}, wdfc, 160, F(IN_DECONV_FC_B),
          bufF, nullptr, 128, 4096, 160, ACT_NONE, stream);
    bnstats(bufF, 128, 4096);
    bn_apply_perm_kernel<<<(128 * 4096 + 255) / 256, 256, 0, stream>>>(
        bufF, bfP, mean, var, F(IN_DBNF_G), F(IN_DBNF_B));
    gemmTap(TapDeconv{bfP, 4, 4, 3, 6}, wd4, F(IN_DECONV4_B),
            bufF, nullptr, 8192, 256, ACT_NONE, stream);
    bnstats(bufF, 8192, 256);
    bn_apply_bf16_kernel<<<(8192 * 256 + 255) / 256, 256, 0, stream>>>(
        bufF, bfQ, 8192 * 256, 255, mean, var, F(IN_DBN4_G), F(IN_DBN4_B));
    gemmTap(TapDeconv{bfQ, 8, 8, 4, 8}, wd3, F(IN_DECONV3_B),
            bufF, nullptr, 32768, 256, ACT_NONE, stream);
    bnstats(bufF, 32768, 256);
    bn_apply_bf16_kernel<<<(32768 * 256 + 255) / 256, 256, 0, stream>>>(
        bufF, bfP, 32768 * 256, 255, mean, var, F(IN_DBN3_G), F(IN_DBN3_B));
    gemmTap(TapDeconv{bfP, 16, 16, 5, 10}, wd2, F(IN_DECONV2_B),
            bufF, nullptr, 131072, 256, ACT_NONE, stream);
    bnstats(bufF, 131072, 256);
    bn_apply_bf16_kernel<<<(131072 * 256 + 255) / 256, 256, 0, stream>>>(
        bufF, bfQ, 131072 * 256, 255, mean, var, F(IN_DBN2_G), F(IN_DBN2_B));
    gemmTap(TapDeconv{bfQ, 32, 32, 6, 12}, wd1, F(IN_DECONV1_B),
            bufF, nullptr, 524288, 3, ACT_NONE, stream);
    recon_kernel<<<(128 * 4096 * 3 + 255) / 256, 256, 0, stream>>>(bufF, rc_o);
}
